// CausalSelfAttention_70248485093913
// MI455X (gfx1250) — matrix operand order repaired and checked by reading, not compiled, not run on hardware
//
#include <hip/hip_runtime.h>

#define DEV_INLINE __device__ __forceinline__

typedef __attribute__((ext_vector_type(8)))  __bf16 bf8v;
typedef __attribute__((ext_vector_type(16))) __bf16 bf16v;
typedef __attribute__((ext_vector_type(8)))  float  f8v;
typedef __attribute__((ext_vector_type(4)))  unsigned int u32x4;
typedef __attribute__((ext_vector_type(8)))  int i32x8;
typedef __attribute__((ext_vector_type(4)))  int i32x4;

#if __has_builtin(__builtin_amdgcn_tensor_load_to_lds)
#define USE_TDM 1
#else
#define USE_TDM 0
#endif

namespace {
constexpr int B_ = 2, T_ = 2048, C_ = 768, H_ = 12, D_ = 64;
constexpr int NQKV = 3 * C_;   // 2304
constexpr int M_   = B_ * T_;  // 4096
constexpr int LDSS = 40;       // LDS row stride (ushorts): 80B = 64B data + 16B pad
constexpr int TILE = 64 * LDSS;
}

DEV_INLINE unsigned short f32_to_bf16(float f) {
  unsigned u = __float_as_uint(f);
  u += 0x7FFFu + ((u >> 16) & 1u);   // round-to-nearest-even
  return (unsigned short)(u >> 16);
}

// A-operand (16x32 MxK): lane m holds 16 contiguous K at 16*hl
DEV_INLINE bf16v make_a(const unsigned short* row, int khalf) {
  bf8v lo = *(const bf8v*)(row + 2 * khalf);
  bf8v hi = *(const bf8v*)(row + 2 * khalf + 8);
  bf16v r;
#pragma unroll
  for (int i = 0; i < 8; ++i) { r[i] = lo[i]; r[i + 8] = hi[i]; }
  return r;
}

// B-operand (32x16 KxN): lane n holds 16 contiguous K at 16*hl
DEV_INLINE bf16v make_b(const unsigned short* p) {
  bf8v lo = *(const bf8v*)p;
  bf8v hi = *(const bf8v*)(p + 8);
  bf16v r;
#pragma unroll
  for (int i = 0; i < 8; ++i) { r[i] = lo[i]; r[i + 8] = hi[i]; }
  return r;
}

DEV_INLINE f8v wmma_bf16(bf16v a, bf16v b, f8v c) {
  return __builtin_amdgcn_wmma_f32_16x16x32_bf16(false, a, false, b, (short)0, c,
                                                 false, false);
}

DEV_INLINE float half_max(float v) {
#pragma unroll
  for (int m = 1; m < 16; m <<= 1) v = fmaxf(v, __shfl_xor(v, m, 32));
  return v;
}
DEV_INLINE float half_sum(float v) {
#pragma unroll
  for (int m = 1; m < 16; m <<= 1) v += __shfl_xor(v, m, 32);
  return v;
}

DEV_INLINE unsigned lds_addr_of(const void* p) {
  return (unsigned)(unsigned long long)p;   // low 32 bits of LDS generic address
}

#if USE_TDM
// Tensor Data Mover: 2-D bf16 tile (tile_d0 x tile_d1) global -> LDS,
// LDS rows padded to LDSS via pad_interval=16 dwords, pad_amount=4 dwords.
DEV_INLINE void tdm_load_2d(unsigned lds_byte_addr, const void* gaddr,
                            int tensor_d0, int tensor_d1, long long stride0,
                            int tile_d0, int tile_d1) {
  unsigned long long ga = (unsigned long long)gaddr;
  u32x4 g0;
  g0[0] = 1u;                                   // count=1 valid user descriptor
  g0[1] = lds_byte_addr;                        // lds_addr
  g0[2] = (unsigned)ga;                         // global_addr[31:0]
  g0[3] = (unsigned)((ga >> 32) & 0x1FFFFFFu)   // global_addr[56:32]
        | 0x80000000u;                          // type = 2 ("image")
  i32x8 g1;
  g1[0] = (1 << 16)    // data_size = 2 bytes
        | (1 << 20)    // pad_enable
        | (3 << 22)    // pad_interval: 16 dwords (one 64B tile row)
        | (3 << 25);   // pad_amount: 4 dwords (16B)
  g1[1] = (tensor_d0 & 0xFFFF) << 16;
  g1[2] = ((tensor_d0 >> 16) & 0xFFFF) | ((tensor_d1 & 0xFFFF) << 16);
  g1[3] = ((tensor_d1 >> 16) & 0xFFFF) | ((tile_d0 & 0xFFFF) << 16);
  g1[4] = (tile_d1 & 0xFFFF);                   // tile_dim1, tile_dim2=0
  g1[5] = (int)(stride0 & 0xFFFFFFFFll);        // tensor_dim0_stride[31:0]
  g1[6] = (int)((stride0 >> 32) & 0xFFFFll);    // stride0[47:32], dim1_stride=0
  g1[7] = 0;
  i32x4 z4 = {0, 0, 0, 0};
  i32x8 z8 = {0, 0, 0, 0, 0, 0, 0, 0};
  // 6-arg variant (clang-23 / therock-10.0): (g0, g1, g2, g3, g4, cpol)
  __builtin_amdgcn_tensor_load_to_lds(g0, g1, z4, z4, z8, 0);
}
#endif

// Shared GEMM inner loop: C[64x64] += A[64xK] * B[64xK]^T, A/B bf16,
// B pre-transposed [N][K]. Ag=&A[m0*K], Bg=&B[n0*K].
DEV_INLINE void gemm_core(const unsigned short* Ag, const unsigned short* Bg,
                          int K, int arows, int brows,
                          unsigned short* As2, unsigned short* Bs2,
                          int tid, f8v acc[4]) {
  const int wave = tid >> 5, lane = tid & 31;
  const int lm = lane & 15, hl = lane >> 4, khalf = hl * 8;
  const int mw = wave * 16;
  const int NK = K / 32;
#if USE_TDM
  if (wave == 0) {
    tdm_load_2d(lds_addr_of(As2), Ag, K, arows, K, 32, 64);
    tdm_load_2d(lds_addr_of(Bs2), Bg, K, brows, K, 32, 64);
  }
  for (int it = 0; it < NK; ++it) {
    const int cur = it & 1;
    unsigned short* Asc = As2 + cur * TILE;
    unsigned short* Bsc = Bs2 + cur * TILE;
    if (wave == 0) {
      if (it + 1 < NK) {   // async-prefetch next k-tile, then wait for current
        tdm_load_2d(lds_addr_of(As2 + (cur ^ 1) * TILE), Ag + (it + 1) * 32,
                    K, arows, K, 32, 64);
        tdm_load_2d(lds_addr_of(Bs2 + (cur ^ 1) * TILE), Bg + (it + 1) * 32,
                    K, brows, K, 32, 64);
        __builtin_amdgcn_s_wait_tensorcnt(2);
      } else {
        __builtin_amdgcn_s_wait_tensorcnt(0);
      }
    }
    __syncthreads();
    bf16v a = make_a(&Asc[(mw + lm) * LDSS], khalf);
#pragma unroll
    for (int j = 0; j < 4; ++j) {
      bf16v bb = make_b(&Bsc[(j * 16 + lm) * LDSS + hl * 16]);
      acc[j] = wmma_bf16(a, bb, acc[j]);
    }
    __syncthreads();
  }
#else
  for (int it = 0; it < NK; ++it) {
    const int row = tid >> 1, ko = (tid & 1) * 16;
    {
      const unsigned short* s = Ag + (size_t)row * K + it * 32 + ko;
      *(uint4*)&As2[row * LDSS + ko]     = *(const uint4*)s;
      *(uint4*)&As2[row * LDSS + ko + 8] = *(const uint4*)(s + 8);
    }
    {
      const unsigned short* s = Bg + (size_t)row * K + it * 32 + ko;
      *(uint4*)&Bs2[row * LDSS + ko]     = *(const uint4*)s;
      *(uint4*)&Bs2[row * LDSS + ko + 8] = *(const uint4*)(s + 8);
    }
    __syncthreads();
    bf16v a = make_a(&As2[(mw + lm) * LDSS], khalf);
#pragma unroll
    for (int j = 0; j < 4; ++j) {
      bf16v bb = make_b(&Bs2[(j * 16 + lm) * LDSS + hl * 16]);
      acc[j] = wmma_bf16(a, bb, acc[j]);
    }
    __syncthreads();
  }
#endif
}

// ---------------------------------------------------------------------------
// Prep: fp32 -> bf16 (straight, vectorized x4)
// ---------------------------------------------------------------------------
__global__ __launch_bounds__(256) void cvt_f32_bf16(
    const float* __restrict__ s, unsigned short* __restrict__ d, int n4) {
  int i = blockIdx.x * blockDim.x + threadIdx.x;
  if (i >= n4) return;
  float4 f = ((const float4*)s)[i];
  uint2 u;
  u.x = (unsigned)f32_to_bf16(f.x) | ((unsigned)f32_to_bf16(f.y) << 16);
  u.y = (unsigned)f32_to_bf16(f.z) | ((unsigned)f32_to_bf16(f.w) << 16);
  ((uint2*)d)[i] = u;
}

// Prep: fp32 [K][N] -> bf16 [N][K] (transpose + convert)
__global__ __launch_bounds__(256) void cvt_transpose(
    const float* __restrict__ src, unsigned short* __restrict__ dst,
    int K, int N) {
  size_t idx = (size_t)blockIdx.x * blockDim.x + threadIdx.x;
  if (idx >= (size_t)K * N) return;
  int n = (int)(idx % N), k = (int)(idx / N);
  dst[(size_t)n * K + k] = f32_to_bf16(src[idx]);
}

// ---------------------------------------------------------------------------
// Kernel 1: qkv = Xb @ Wqt^T + b, scatter Q,K -> [B,H,T,D], V -> [B,H,D,T]
// ---------------------------------------------------------------------------
__global__ __launch_bounds__(128) void qkv_gemm(
    const unsigned short* __restrict__ Xb, const unsigned short* __restrict__ Wqt,
    const float* __restrict__ bq, unsigned short* __restrict__ Qb,
    unsigned short* __restrict__ Kb, unsigned short* __restrict__ Vt) {
  __shared__ __align__(16) unsigned short As2[2 * TILE];
  __shared__ __align__(16) unsigned short Bs2[2 * TILE];
  const int tid = threadIdx.x;
  const int wave = tid >> 5, lane = tid & 31;
  const int lm = lane & 15, hl = lane >> 4;
  const int m0 = blockIdx.x * 64, n0 = blockIdx.y * 64, mw = wave * 16;

  f8v acc[4] = {};
  gemm_core(Xb + (size_t)m0 * C_, Wqt + (size_t)n0 * C_, C_, M_, NQKV,
            As2, Bs2, tid, acc);

#pragma unroll
  for (int j = 0; j < 4; ++j) {
    int gc = n0 + j * 16 + lm;
    int sec = gc / C_, rem = gc % C_;
    int h = rem >> 6, d = rem & 63;
    float bias = bq[gc];
#pragma unroll
    for (int r = 0; r < 8; ++r) {
      int gm = m0 + mw + r + 8 * hl;
      int b = gm >> 11, t = gm & (T_ - 1);
      int bh = b * H_ + h;
      unsigned short v = f32_to_bf16(acc[j][r] + bias);
      if (sec == 0)      Qb[((size_t)bh * T_ + t) * D_ + d] = v;
      else if (sec == 1) Kb[((size_t)bh * T_ + t) * D_ + d] = v;
      else               Vt[((size_t)bh * D_ + d) * T_ + t] = v;  // transposed
    }
  }
}

// ---------------------------------------------------------------------------
// Kernel 2: flash attention. 64 q-rows/block (4 waves x 16), 32-key tiles,
// online softmax. V read directly from transposed [B,H,D,T] layout: no
// cooperative staging, no block barriers. Output bf16 [B*T, C].
// ---------------------------------------------------------------------------
__global__ __launch_bounds__(128) void attn_fwd(
    const unsigned short* __restrict__ Qb, const unsigned short* __restrict__ Kb,
    const unsigned short* __restrict__ Vt, unsigned short* __restrict__ O) {
  __shared__ __align__(16) unsigned short Ps[4 * 16 * LDSS];  // per-wave P tile
  const int tid  = threadIdx.x;
  const int wave = tid >> 5, lane = tid & 31;
  const int lm = lane & 15, hl = lane >> 4, khalf = hl * 8;
  const int q0 = blockIdx.x * 64;
  const int bh = blockIdx.y;
  const unsigned short* Qh = Qb + (size_t)bh * T_ * D_;
  const unsigned short* Kh = Kb + (size_t)bh * T_ * D_;
  const unsigned short* Vh = Vt + (size_t)bh * D_ * T_;
  const int qw = q0 + wave * 16;

  const unsigned short* qrow = Qh + (size_t)(qw + lm) * D_;
  bf16v aq0 = make_a(qrow, khalf);
  bf16v aq1 = make_a(qrow + 32, khalf);

  f8v o[4] = {};
  float mrow[8], lrow[8];
#pragma unroll
  for (int r = 0; r < 8; ++r) { mrow[r] = -__builtin_inff(); lrow[r] = 0.0f; }

  const int ntiles = (q0 + 64) / 32;
  for (int kt = 0; kt < ntiles; ++kt) {
    const int kb = kt * 32;
    if (kt + 1 < ntiles) {  // prefetch next K/V tile into cache
      __builtin_prefetch(Kh + (size_t)(kb + 32 + lane) * D_, 0, 1);
      __builtin_prefetch(Vh + (size_t)lane * T_ + kb + 32, 0, 1);
      __builtin_prefetch(Vh + (size_t)(lane + 32) * T_ + kb + 32, 0, 1);
    }

    // S = Q @ K^T
    f8v s[2];
#pragma unroll
    for (int sub = 0; sub < 2; ++sub) {
      const unsigned short* krow = Kh + (size_t)(kb + sub * 16 + lm) * D_;
      bf16v bk0 = make_b(krow + hl * 16);
      bf16v bk1 = make_b(krow + 32 + hl * 16);
      f8v a = {};
      a = wmma_bf16(aq0, bk0, a);
      a = wmma_bf16(aq1, bk1, a);
      s[sub] = a;
    }

    const float scale = 0.125f;  // 1/sqrt(64)
#pragma unroll
    for (int r = 0; r < 8; ++r) {
      const int q  = qw + r + 8 * hl;
      const int mq = r + 8 * hl;
      float s0 = s[0][r] * scale;
      float s1 = s[1][r] * scale;
      s0 = (kb + lm      > q) ? -__builtin_inff() : s0;
      s1 = (kb + 16 + lm > q) ? -__builtin_inff() : s1;
      float tm    = half_max(fmaxf(s0, s1));
      float newm  = fmaxf(mrow[r], tm);
      float alpha = __expf(mrow[r] - newm);
      float p0 = __expf(s0 - newm);
      float p1 = __expf(s1 - newm);
      lrow[r] = lrow[r] * alpha + half_sum(p0 + p1);
      mrow[r] = newm;
#pragma unroll
      for (int j = 0; j < 4; ++j) o[j][r] *= alpha;
      unsigned short* pr = &Ps[(wave * 16 + mq) * LDSS];
      pr[lm]      = f32_to_bf16(p0);
      pr[16 + lm] = f32_to_bf16(p1);
    }

    // O += P @ V  (per-wave LDS round trip: C-layout P -> A-layout)
    {
      const unsigned short* pr = &Ps[(wave * 16 + lm) * LDSS];
      bf16v pa = make_a(pr, khalf);
#pragma unroll
      for (int j = 0; j < 4; ++j) {
        bf16v bv = make_b(Vh + (size_t)(j * 16 + lm) * T_ + kb + hl * 16);
        o[j] = wmma_bf16(pa, bv, o[j]);
      }
    }
  }

  const int b = bh / H_, h = bh % H_;
#pragma unroll
  for (int r = 0; r < 8; ++r) {
    float inv = 1.0f / lrow[r];
    int q = qw + r + 8 * hl;
    size_t base = ((size_t)(b * T_ + q)) * C_ + h * D_;
#pragma unroll
    for (int j = 0; j < 4; ++j)
      O[base + j * 16 + lm] = f32_to_bf16(o[j][r] * inv);
  }
}

// ---------------------------------------------------------------------------
// Kernel 3: out = AO @ Wpt^T + b_proj (fp32 output)
// ---------------------------------------------------------------------------
__global__ __launch_bounds__(128) void proj_gemm(
    const unsigned short* __restrict__ AO, const unsigned short* __restrict__ Wpt,
    const float* __restrict__ bp, float* __restrict__ out) {
  __shared__ __align__(16) unsigned short As2[2 * TILE];
  __shared__ __align__(16) unsigned short Bs2[2 * TILE];
  const int tid = threadIdx.x;
  const int wave = tid >> 5, lane = tid & 31;
  const int lm = lane & 15, hl = lane >> 4;
  const int m0 = blockIdx.x * 64, n0 = blockIdx.y * 64, mw = wave * 16;

  f8v acc[4] = {};
  gemm_core(AO + (size_t)m0 * C_, Wpt + (size_t)n0 * C_, C_, M_, C_,
            As2, Bs2, tid, acc);

#pragma unroll
  for (int j = 0; j < 4; ++j) {
    int gc = n0 + j * 16 + lm;
    float bias = bp[gc];
#pragma unroll
    for (int r = 0; r < 8; ++r) {
      int gm = m0 + mw + r + 8 * hl;
      out[(size_t)gm * C_ + gc] = acc[j][r] + bias;
    }
  }
}

// ---------------------------------------------------------------------------
extern "C" void kernel_launch(void* const* d_in, const int* in_sizes, int n_in,
                              void* d_out, int out_size, void* d_ws, size_t ws_size,
                              hipStream_t stream) {
  const float* x  = (const float*)d_in[0];
  const float* Wq = (const float*)d_in[1];
  const float* bq = (const float*)d_in[2];
  const float* Wp = (const float*)d_in[3];
  const float* bp = (const float*)d_in[4];
  float* out = (float*)d_out;

  char* ws = (char*)d_ws;
  const size_t headB = (size_t)B_ * H_ * T_ * D_ * 2;  // 6,291,456
  unsigned short* Qb  = (unsigned short*)(ws);
  unsigned short* Kb  = (unsigned short*)(ws + headB);
  unsigned short* Vt  = (unsigned short*)(ws + 2 * headB);
  unsigned short* AO  = (unsigned short*)(ws + 3 * headB);
  unsigned short* Xb  = (unsigned short*)(ws + 4 * headB);
  unsigned short* Wqt = (unsigned short*)(ws + 5 * headB);
  unsigned short* Wpt = (unsigned short*)(ws + 5 * headB + (size_t)NQKV * C_ * 2);

  // one-time bf16 conversion / weight transpose
  cvt_f32_bf16 <<<dim3((M_ * C_ / 4 + 255) / 256), 256, 0, stream>>>(x, Xb, M_ * C_ / 4);
  cvt_transpose<<<dim3((C_ * NQKV + 255) / 256), 256, 0, stream>>>(Wq, Wqt, C_, NQKV);
  cvt_transpose<<<dim3((C_ * C_ + 255) / 256), 256, 0, stream>>>(Wp, Wpt, C_, C_);

  qkv_gemm <<<dim3(M_ / 64, NQKV / 64), 128, 0, stream>>>(Xb, Wqt, bq, Qb, Kb, Vt);
  attn_fwd <<<dim3(T_ / 64, B_ * H_), 128, 0, stream>>>(Qb, Kb, Vt, AO);
  proj_gemm<<<dim3(M_ / 64, C_ / 64), 128, 0, stream>>>(AO, Wpt, bp, out);
}
